// MultiHeadAttentionBlock_2370821947846
// MI455X (gfx1250) — matrix operand order repaired and checked by reading, not compiled, not run on hardware
//
#include <hip/hip_runtime.h>
#include <math.h>

// ---------------------------------------------------------------------------
// MultiHeadAttention block for MI455X (gfx1250, wave32, WMMA).
// Pipeline: 3x GEMM (q/k/v proj, bf16 WMMA; V stored head-transposed) ->
// flash attention (bf16 WMMA, fp32 online softmax, row-sum via WMMA-with-ones,
// row-max via DPP row_xmask, async double-buffered K/V tiles) ->
// GEMM (output proj, fp32 out).
// ---------------------------------------------------------------------------

typedef __attribute__((ext_vector_type(16))) __bf16 v16bf;
typedef __attribute__((ext_vector_type(2)))  __bf16 v2bf;
typedef __attribute__((ext_vector_type(8)))  float  v8f;
typedef __attribute__((ext_vector_type(4)))  unsigned int u32x4;

// int4 vector matching the async-to-LDS builtin's parameter type
typedef int i32x4v __attribute__((vector_size(16)));
typedef __attribute__((address_space(1))) i32x4v as1_i32x4;
typedef __attribute__((address_space(3))) i32x4v as3_i32x4;

static constexpr int Bc = 4, Sc = 2048, Dc = 1024, Hc = 16, Dh = 64;

// ---------------- fp32 -> bf16 conversion ----------------------------------
// Prefer the packed hw builtin when the toolchain exposes it; otherwise use
// native __bf16 fptrunc and let the backend pick the best gfx1250 lowering.
#if defined(__has_builtin)
#if __has_builtin(__builtin_amdgcn_cvt_pk_bf16_f32)
#define USE_PK_CVT 1
#endif
#endif
#ifndef USE_PK_CVT
#define USE_PK_CVT 0
#endif

// two floats -> packed 2x bf16 in one dword (lo in [15:0], hi in [31:16])
__device__ __forceinline__ unsigned cvt2bf(float lo, float hi) {
    union { v2bf v; unsigned u; } c;
#if USE_PK_CVT
    c.v = __builtin_amdgcn_cvt_pk_bf16_f32(lo, hi);
#else
    c.v[0] = (__bf16)lo;
    c.v[1] = (__bf16)hi;
#endif
    return c.u;
}
__device__ __forceinline__ unsigned short f32_bf16(float f) {
    union { __bf16 h; unsigned short u; } c;
    c.h = (__bf16)f;
    return c.u;
}

union FragBF {                 // one WMMA 16-bf16 per-lane fragment (8 VGPRs)
    v16bf v;
    u32x4 q[2];
    unsigned u[8];
    unsigned short s[16];
};

union Pack8 {                  // 8 bf16 = 16 bytes staging packet
    unsigned short s[8];
    unsigned u[4];
    u32x4 v;
};

// ---------------- async global->LDS copy (16B per lane), guarded -----------
#if defined(__has_builtin)
#if __has_builtin(__builtin_amdgcn_global_load_async_to_lds_b128) && \
    __has_builtin(__builtin_amdgcn_s_wait_asynccnt)
#define USE_ASYNC_LDS 1
#endif
#endif
#ifndef USE_ASYNC_LDS
#define USE_ASYNC_LDS 0
#endif

__device__ __forceinline__ void copy16_g2l(const unsigned short* g, unsigned short* l) {
#if USE_ASYNC_LDS
    as1_i32x4* gp = (as1_i32x4*)(void*)const_cast<unsigned short*>(g);
    as3_i32x4* lp = (as3_i32x4*)(void*)l;
    __builtin_amdgcn_global_load_async_to_lds_b128(gp, lp, 0, 0);
#else
    *(u32x4*)l = *(const u32x4*)g;
#endif
}
__device__ __forceinline__ void async_drain() {
#if USE_ASYNC_LDS
    __builtin_amdgcn_s_wait_asynccnt(0);
#endif
}

// ---------------- row max over 16-lane half via DPP row_xmask --------------
#if defined(__has_builtin)
#if __has_builtin(__builtin_amdgcn_update_dpp)
#define USE_DPP_RED 1
#endif
#endif
#ifndef USE_DPP_RED
#define USE_DPP_RED 0
#endif

__device__ __forceinline__ float rowmax16(float x) {
#if USE_DPP_RED
#define DPP_XMAX(CTRL)                                                         \
    x = fmaxf(x, __int_as_float(__builtin_amdgcn_update_dpp(                   \
            0, __float_as_int(x), (CTRL), 0xf, 0xf, true)))
    DPP_XMAX(0x161);   // row_xmask:1
    DPP_XMAX(0x162);   // row_xmask:2
    DPP_XMAX(0x164);   // row_xmask:4
    DPP_XMAX(0x168);   // row_xmask:8
#undef DPP_XMAX
#else
    #pragma unroll
    for (int off = 1; off < 16; off <<= 1)
        x = fmaxf(x, __shfl_xor(x, off, 32));
#endif
    return x;
}

// ---------------------------------------------------------------------------
// Tiled GEMM:  out[m,n] = out_scale * sum_k A[m,k] * W[n,k]
// A: fp32 or bf16 [M,K]; W: fp32 [N,K].
// OUT_MODE 0: fp32 [M,N]     (final projection -> d_out)
// OUT_MODE 1: bf16 [B,H,S,Dh] head-split        (Q, K)
// OUT_MODE 2: bf16 [B,H,Dh,S] head-transposed   (V)
// Block 256 thr = 8 waves; tile BM=64 x BN=128 x BK=32; wave tile 32x32.
// ---------------------------------------------------------------------------
template <bool ABF16, int OUT_MODE>
__global__ __launch_bounds__(256)
void gemm_wmma_kernel(const void* __restrict__ Av, const float* __restrict__ W,
                      void* __restrict__ Ov, int M, int N, int K, float out_scale)
{
    constexpr int BM = 64, BN = 128, BK = 32;
    constexpr int LDA = 40, LDB = 40;          // ushort strides: 80B, 16B-aligned, bank-spread
    __shared__ unsigned short sA[BM * LDA];
    __shared__ unsigned short sB[BN * LDB];

    const int tid  = threadIdx.x;
    const int lane = tid & 31, wave = tid >> 5;
    const int kh   = lane >> 4, ln = lane & 15;
    const int wm   = (wave >> 2) * 32;
    const int wn   = (wave & 3) * 32;
    const int m0   = blockIdx.y * BM, n0 = blockIdx.x * BN;

    v8f acc[2][2] = {};

    for (int k0 = 0; k0 < K; k0 += BK) {
        __syncthreads();
        // ---- stage A tile (64x32): 256 chunks of 8, 1/thread ----
        {
            const int row = tid >> 2, c8 = tid & 3;
            const int gk  = k0 + c8 * 8;
            Pack8 t;
            if (ABF16) {
                const unsigned short* Ab = (const unsigned short*)Av;
                t.v = *(const u32x4*)(Ab + (size_t)(m0 + row) * K + gk);
                if (gk + BK < K) __builtin_prefetch(Ab + (size_t)(m0 + row) * K + gk + BK, 0, 1);
            } else {
                const float* p = (const float*)Av + (size_t)(m0 + row) * K + gk;
                #pragma unroll
                for (int e = 0; e < 4; ++e) t.u[e] = cvt2bf(p[2 * e], p[2 * e + 1]);
                if (gk + BK < K) __builtin_prefetch(p + BK, 0, 1);
            }
            *(u32x4*)(sA + row * LDA + c8 * 8) = t.v;
        }
        // ---- stage B tile (128x32 of W rows): 512 chunks, 2/thread ----
        #pragma unroll
        for (int it = 0; it < 2; ++it) {
            const int idx = tid + it * 256;
            const int row = idx >> 2, c8 = idx & 3;
            const float* p = W + (size_t)(n0 + row) * K + k0 + c8 * 8;
            Pack8 t;
            #pragma unroll
            for (int e = 0; e < 4; ++e) t.u[e] = cvt2bf(p[2 * e], p[2 * e + 1]);
            if (k0 + BK < K) __builtin_prefetch(p + BK, 0, 1);
            *(u32x4*)(sB + row * LDB + c8 * 8) = t.v;
        }
        __syncthreads();

        FragBF a[2], b[2];
        #pragma unroll
        for (int i = 0; i < 2; ++i) {          // A 16x32: chunks at K = 8*kh, 16+8*kh
            const int row = wm + i * 16 + ln;
            a[i].q[0] = *(const u32x4*)(sA + row * LDA + 8 * kh);
            a[i].q[1] = *(const u32x4*)(sA + row * LDA + 16 + 8 * kh);
        }
        #pragma unroll
        for (int j = 0; j < 2; ++j) {          // B 32x16: chunks at K = 8*kh, 16+8*kh
            const int col = wn + j * 16 + ln;
            b[j].q[0] = *(const u32x4*)(sB + col * LDB + 8 * kh);
            b[j].q[1] = *(const u32x4*)(sB + col * LDB + 16 + 8 * kh);
        }
        #pragma unroll
        for (int i = 0; i < 2; ++i)
            #pragma unroll
            for (int j = 0; j < 2; ++j)
                acc[i][j] = __builtin_amdgcn_wmma_f32_16x16x32_bf16(
                    false, a[i].v, false, b[j].v, (short)0, acc[i][j], false, false);
    }

    // ---- epilogue: C layout (N = ln, M = r + 8*kh) ----
    #pragma unroll
    for (int i = 0; i < 2; ++i)
        #pragma unroll
        for (int j = 0; j < 2; ++j) {
            const int n = n0 + wn + j * 16 + ln;
            if (OUT_MODE == 2) {
                // bf16 [B,H,Dh,S]; consecutive r -> consecutive s: pack pairs
                unsigned short* Ob = (unsigned short*)Ov;
                #pragma unroll
                for (int r = 0; r < 8; r += 2) {
                    const int m  = m0 + wm + i * 16 + r + 8 * kh;
                    const int bb = m >> 11, s = m & (Sc - 1);
                    const int h  = n >> 6,  d = n & (Dh - 1);
                    *(unsigned int*)(Ob + (((size_t)(bb * Hc + h)) * Dh + d) * Sc + s)
                        = cvt2bf(acc[i][j][r] * out_scale, acc[i][j][r + 1] * out_scale);
                }
            } else {
                #pragma unroll
                for (int r = 0; r < 8; ++r) {
                    const int m   = m0 + wm + i * 16 + r + 8 * kh;
                    const float v = acc[i][j][r] * out_scale;
                    if (OUT_MODE == 1) {
                        unsigned short* Ob = (unsigned short*)Ov;
                        const int bb = m >> 11, s = m & (Sc - 1);
                        const int h  = n >> 6,  d = n & (Dh - 1);
                        Ob[(((size_t)(bb * Hc + h)) * Sc + s) * Dh + d] = f32_bf16(v);
                    } else {
                        ((float*)Ov)[(size_t)m * N + n] = v;
                    }
                }
            }
        }
}

// ---------------------------------------------------------------------------
// Flash attention per (b,h): wave = 16 query rows, block = 8 waves = 128 rows.
// K tiles [32 keys][64 d], V tiles transposed [64 d][32 keys], both async
// double-buffered in LDS. Scores via WMMA; row-max via DPP; row-sum via an
// extra WMMA against an all-ones B fragment; P round-trips per-wave LDS to
// convert C-layout -> A-layout. Scale 1/sqrt(Dh) pre-folded into Q.
// ---------------------------------------------------------------------------
__global__ __launch_bounds__(256)
void attention_kernel(const unsigned short* __restrict__ Qb,
                      const unsigned short* __restrict__ Kb,
                      const unsigned short* __restrict__ Vt,
                      const int* __restrict__ mask,
                      unsigned short* __restrict__ Ob)
{
    constexpr int LKV = 72;   // K-tile ushort stride (144B, 16B-aligned, bank-spread)
    constexpr int LDT = 40;   // V-tile ushort stride (80B)
    constexpr int LPP = 40;   // P-tile ushort stride (80B)
    __shared__ unsigned short sK[2][32 * LKV];
    __shared__ unsigned short sVt[2][Dh * LDT];
    __shared__ unsigned short sP[8 * 16 * LPP];

    const int tid = threadIdx.x, lane = tid & 31, wave = tid >> 5;
    const int kh = lane >> 4, ln = lane & 15;
    const int bh = blockIdx.y;
    const int b  = bh >> 4, h = bh & (Hc - 1);
    const int rowbase = blockIdx.x * 128 + wave * 16;

    const unsigned short* kbase  = Kb + (size_t)bh * Sc * Dh;
    const unsigned short* vtbase = Vt + (size_t)bh * Dh * Sc;

    // per-thread staging coordinates
    const int krow = tid >> 3, kc8 = tid & 7;   // K tile: 32 rows x 8 chunks
    const int vrow = tid >> 2, vc8 = tid & 3;   // V tile: 64 rows x 4 chunks

    // Q fragments (16 rows x Dh=64 -> two 16x32 A frags), scale pre-folded.
    FragBF qf[2];
    {
        const unsigned short* qp = Qb + ((size_t)bh * Sc + rowbase + ln) * Dh;
        #pragma unroll
        for (int t = 0; t < 2; ++t) {
            qf[t].q[0] = *(const u32x4*)(qp + 32 * t + 8 * kh);
            qf[t].q[1] = *(const u32x4*)(qp + 32 * t + 16 + 8 * kh);
        }
    }

    FragBF onesf;                       // all-ones B fragment for row sums
    #pragma unroll
    for (int w = 0; w < 8; ++w) onesf.u[w] = 0x3f803f80u;   // bf16(1.0) x2

    v8f   o[4] = {}, lacc = {};
    float mrow[8];
    #pragma unroll
    for (int r = 0; r < 8; ++r) mrow[r] = -__builtin_inff();

    // prologue: issue tile 0 into buffer 0
    copy16_g2l(kbase  + (size_t)krow * Dh + kc8 * 8, &sK[0][krow * LKV + kc8 * 8]);
    copy16_g2l(vtbase + (size_t)vrow * Sc + vc8 * 8, &sVt[0][vrow * LDT + vc8 * 8]);

    const int NKB = Sc / 32;
    for (int i = 0; i < NKB; ++i) {
        const int kb  = i * 32;
        const int cur = i & 1, nxt = cur ^ 1;

        async_drain();                 // this wave's async copies have landed
        __syncthreads();               // everyone's copies visible; prev readers done

        if (i + 1 < NKB) {             // software pipeline: fetch next tile
            const int kn = kb + 32;
            copy16_g2l(kbase  + (size_t)(kn + krow) * Dh + kc8 * 8,
                       &sK[nxt][krow * LKV + kc8 * 8]);
            copy16_g2l(vtbase + (size_t)vrow * Sc + kn + vc8 * 8,
                       &sVt[nxt][vrow * LDT + vc8 * 8]);
        }

        // ---- scores: 16 rows x 32 keys = two C frags, 2 K-steps over Dh ----
        v8f sc[2] = {};
        #pragma unroll
        for (int j = 0; j < 2; ++j)
            #pragma unroll
            for (int ks = 0; ks < 2; ++ks) {
                FragBF bk;
                const unsigned short* kp = &sK[cur][(j * 16 + ln) * LKV + ks * 32 + 8 * kh];
                bk.q[0] = *(const u32x4*)(kp);
                bk.q[1] = *(const u32x4*)(kp + 16);
                sc[j] = __builtin_amdgcn_wmma_f32_16x16x32_bf16(
                    false, qf[ks].v, false, bk.v, (short)0, sc[j], false, false);
            }

        // ---- mask ----
        const int sg = rowbase + 8 * kh;
        #pragma unroll
        for (int j = 0; j < 2; ++j) {
            const int tg = kb + j * 16 + ln;
            const int* mp = mask + ((size_t)b * Sc + sg) * Sc + tg;
            #pragma unroll
            for (int r = 0; r < 8; ++r)
                if (mp[(size_t)r * Sc] == 0) sc[j][r] = -__builtin_inff();
        }

        // ---- online softmax: max via DPP, sum via WMMA later ----
        float alpha[8];
        #pragma unroll
        for (int r = 0; r < 8; ++r) {
            const float rm   = rowmax16(fmaxf(sc[0][r], sc[1][r]));
            const float mnew = fmaxf(mrow[r], rm);
            alpha[r] = __expf(mrow[r] - mnew);
            sc[0][r] = __expf(sc[0][r] - mnew);
            sc[1][r] = __expf(sc[1][r] - mnew);
            mrow[r]  = mnew;
        }
        #pragma unroll
        for (int j = 0; j < 4; ++j)
            #pragma unroll
            for (int r = 0; r < 8; ++r)
                o[j][r] *= alpha[r];
        #pragma unroll
        for (int r = 0; r < 8; ++r)
            lacc[r] *= alpha[r];

        // ---- P: C-layout regs -> per-wave LDS tile (row-major 16x32) ----
        unsigned short* pw = sP + wave * 16 * LPP;
        #pragma unroll
        for (int j = 0; j < 2; ++j)
            #pragma unroll
            for (int r = 0; r < 8; ++r)
                pw[(r + 8 * kh) * LPP + j * 16 + ln] = f32_bf16(sc[j][r]);

        __syncthreads();

        // ---- O += P @ V ; l += P @ 1  (K = 32 keys) ----
        FragBF pa;
        pa.q[0] = *(const u32x4*)(pw + ln * LPP + 8 * kh);
        pa.q[1] = *(const u32x4*)(pw + ln * LPP + 16 + 8 * kh);
        #pragma unroll
        for (int j = 0; j < 4; ++j) {
            FragBF bv;                          // V^T: keys 8*kh..+7 and 16+8*kh..+7 per lane
            const unsigned short* vp = &sVt[cur][(j * 16 + ln) * LDT + 8 * kh];
            bv.q[0] = *(const u32x4*)(vp);
            bv.q[1] = *(const u32x4*)(vp + 16);
            o[j] = __builtin_amdgcn_wmma_f32_16x16x32_bf16(
                false, pa.v, false, bv.v, (short)0, o[j], false, false);
        }
        lacc = __builtin_amdgcn_wmma_f32_16x16x32_bf16(
            false, pa.v, false, onesf.v, (short)0, lacc, false, false);
    }

    // ---- normalize + write concat-heads bf16 [B,S,D] ----
    #pragma unroll
    for (int r = 0; r < 8; ++r) {
        const float inv = 1.0f / lacc[r];
        const int s = rowbase + r + 8 * kh;
        #pragma unroll
        for (int j = 0; j < 4; ++j) {
            const int d = h * Dh + j * 16 + ln;
            Ob[((size_t)b * Sc + s) * Dc + d] = f32_bf16(o[j][r] * inv);
        }
    }
}

// ---------------------------------------------------------------------------
extern "C" void kernel_launch(void* const* d_in, const int* in_sizes, int n_in,
                              void* d_out, int out_size, void* d_ws, size_t ws_size,
                              hipStream_t stream)
{
    const float* query = (const float*)d_in[0];
    const float* key   = (const float*)d_in[1];
    const float* value = (const float*)d_in[2];
    const int*   mask  = (const int*)d_in[3];
    const float* w_q   = (const float*)d_in[4];
    const float* w_k   = (const float*)d_in[5];
    const float* w_v   = (const float*)d_in[6];
    const float* w_o   = (const float*)d_in[7];

    const int M = Bc * Sc;     // 8192
    const int N = Dc, K = Dc;  // 1024

    const size_t elems = (size_t)M * Dc;   // 8.4M bf16 per buffer
    unsigned short* qb    = (unsigned short*)d_ws;
    unsigned short* kbuf  = qb + elems;
    unsigned short* vbuf  = qb + 2 * elems;
    unsigned short* attnb = qb + 3 * elems;

    dim3 blk(256);
    dim3 ggrid(N / 128, M / 64);           // 8 x 128 blocks

    // fold 1/sqrt(Dh) into Q projection
    gemm_wmma_kernel<false, 1><<<ggrid, blk, 0, stream>>>(query, w_q, qb,   M, N, K, 0.125f);
    gemm_wmma_kernel<false, 1><<<ggrid, blk, 0, stream>>>(key,   w_k, kbuf, M, N, K, 1.0f);
    gemm_wmma_kernel<false, 2><<<ggrid, blk, 0, stream>>>(value, w_v, vbuf, M, N, K, 1.0f);

    dim3 agrid(Sc / 128, Bc * Hc);         // 16 x 64 blocks
    attention_kernel<<<agrid, blk, 0, stream>>>(qb, kbuf, vbuf, mask, attnb);

    gemm_wmma_kernel<true, 0><<<ggrid, blk, 0, stream>>>(attnb, w_o, d_out, M, N, K, 1.0f);
}
